// HDRDecoderV2_14955076124842
// MI455X (gfx1250) — matrix operand order repaired and checked by reading, not compiled, not run on hardware
//
#include <hip/hip_runtime.h>
#include <hip/hip_bf16.h>

typedef __attribute__((ext_vector_type(16))) _Float16 v16h;
typedef __attribute__((ext_vector_type(8)))  float    v8f;

// ---- WMMA fragment index maps (16-bit data, wave32) ----
// A (16x32, MxK): VGPR v holds K pairs; lanes 16-31 offset K by +8 (first 4 VGPRs)
__device__ __forceinline__ int kmapA(int h, int lane) {
  int v = h >> 1, p = h & 1;
  int base = (v < 4) ? (2 * v + p) : (16 + 2 * (v - 4) + p);
  return base + ((lane & 16) ? 8 : 0);
}
// B (32x16, KxN): lane = column; same K order as A (lanes 16-31 offset K by +8)
__device__ __forceinline__ int kmapB(int h, int lane) {
  return ((h < 8) ? h : (16 + (h - 8))) + ((lane & 16) ? 8 : 0);
}

// ================= conv1: (N,4,64,64) -> relu -> (N,16,32,32), 3x3 s2 p1 ====
__global__ __launch_bounds__(256)
void conv1_kernel(const float* __restrict__ z, const float* __restrict__ w1,
                  const float* __restrict__ b1, float* __restrict__ x1, int total) {
  int idx = blockIdx.x * 256 + threadIdx.x;
  if (idx >= total) return;
  int x = idx & 31, y = (idx >> 5) & 31, oc = (idx >> 10) & 15, n = idx >> 14;
  float acc = b1[oc];
  #pragma unroll
  for (int ic = 0; ic < 4; ++ic) {
    const float* zin = z + ((n * 4 + ic) * 64) * 64;
    const float* wk  = w1 + (oc * 4 + ic) * 9;
    #pragma unroll
    for (int ky = 0; ky < 3; ++ky) {
      int iy = y * 2 - 1 + ky;
      if (iy < 0 || iy >= 64) continue;
      #pragma unroll
      for (int kx = 0; kx < 3; ++kx) {
        int ix = x * 2 - 1 + kx;
        if (ix < 0 || ix >= 64) continue;
        acc = fmaf(zin[iy * 64 + ix], wk[ky * 3 + kx], acc);
      }
    }
  }
  x1[idx] = fmaxf(acc, 0.0f);
}

// ================= conv2: (N,16,32,32) -> relu -> (N,32,16,16), 3x3 s2 p1 ===
__global__ __launch_bounds__(256)
void conv2_kernel(const float* __restrict__ x1, const float* __restrict__ w2,
                  const float* __restrict__ b2, float* __restrict__ x2, int total) {
  int idx = blockIdx.x * 256 + threadIdx.x;
  if (idx >= total) return;
  int x = idx & 15, y = (idx >> 4) & 15, oc = (idx >> 8) & 31, n = idx >> 13;
  float acc = b2[oc];
  for (int ic = 0; ic < 16; ++ic) {
    const float* xin = x1 + ((n * 16 + ic) * 32) * 32;
    const float* wk  = w2 + (oc * 16 + ic) * 9;
    #pragma unroll
    for (int ky = 0; ky < 3; ++ky) {
      int iy = y * 2 - 1 + ky;
      if (iy < 0 || iy >= 32) continue;
      #pragma unroll
      for (int kx = 0; kx < 3; ++kx) {
        int ix = x * 2 - 1 + kx;
        if (ix < 0 || ix >= 32) continue;
        acc = fmaf(xin[iy * 32 + ix], wk[ky * 3 + kx], acc);
      }
    }
  }
  x2[idx] = fmaxf(acc, 0.0f);
}

// ===== conv3 (1x1, 32->64) + conv4 (1x1, 64->96) fused as WMMA GEMMs ========
// Rows = N*256 pixels (16x16 spatial, per batch). One wave per 16-row M tile.
// Output written permuted + fp16: gridh[((n*8+z)*16+y)*16+x)*12 + cd]
__global__ __launch_bounds__(32)
void coeff_gemm_kernel(const float* __restrict__ x2,   // (N,32,16,16)
                       const float* __restrict__ w3,   // (64,32)
                       const float* __restrict__ b3,   // (64)
                       const float* __restrict__ w4,   // (96,64)
                       const float* __restrict__ b4,   // (96)
                       _Float16* __restrict__ gridh) {
  __shared__ _Float16 Y[16 * 64];                      // conv3 output tile
  const int lane = threadIdx.x;
  const int m0   = blockIdx.x * 16;
  const int mrow = lane & 15;
  const int hi   = (lane & 16) ? 1 : 0;

  // ---- A fragment for conv3: rows m0..m0+15 of X (N*256 x 32) ----
  int r = m0 + mrow;
  int n = r >> 8, s = r & 255;
  int py = s >> 4, px = s & 15;
  const float* xbase = x2 + ((n * 32) * 16 + py) * 16 + px;   // channel stride 256
  v16h a;
  #pragma unroll
  for (int h = 0; h < 16; ++h) a[h] = (_Float16)xbase[kmapA(h, lane) * 256];

  // ---- conv3: Y(16x64) = A(16x32) * W3^T(32x64) + b3 ----
  #pragma unroll
  for (int nt = 0; nt < 4; ++nt) {
    const int col = nt * 16 + mrow;
    const float* wrow = w3 + col * 32;
    v16h b;
    #pragma unroll
    for (int h = 0; h < 16; ++h) b[h] = (_Float16)wrow[kmapB(h, lane)];
    v8f c = {};
    c = __builtin_amdgcn_wmma_f32_16x16x32_f16(false, a, false, b, (short)0, c,
                                               false, false);
    const float bias = b3[col];
    #pragma unroll
    for (int j = 0; j < 8; ++j)
      Y[(j + 8 * hi) * 64 + col] = (_Float16)(c[j] + bias);
  }
  __syncthreads();

  // ---- A fragments for conv4 (K = 64 -> two 16x32 fragments) ----
  v16h a0, a1;
  #pragma unroll
  for (int h = 0; h < 16; ++h) {
    int k = kmapA(h, lane);
    a0[h] = Y[mrow * 64 + k];
    a1[h] = Y[mrow * 64 + k + 32];
  }

  // ---- conv4: G(16x96) = Y(16x64) * W4^T(64x96) + b4 ----
  #pragma unroll
  for (int nt = 0; nt < 6; ++nt) {
    const int col = nt * 16 + mrow;                     // channel 0..95
    const float* wrow = w4 + col * 64;
    v16h b0, b1;
    #pragma unroll
    for (int h = 0; h < 16; ++h) {
      int k = kmapB(h, lane);
      b0[h] = (_Float16)wrow[k];
      b1[h] = (_Float16)wrow[k + 32];
    }
    v8f c = {};
    c = __builtin_amdgcn_wmma_f32_16x16x32_f16(false, a0, false, b0, (short)0, c,
                                               false, false);
    c = __builtin_amdgcn_wmma_f32_16x16x32_f16(false, a1, false, b1, (short)0, c,
                                               false, false);
    const float bias = b4[col];
    const int cd = col >> 3, zz = col & 7;              // reshape (96)->(12,8)
    #pragma unroll
    for (int j = 0; j < 8; ++j) {
      int row = m0 + j + 8 * hi;
      int nn = row >> 8, ss = row & 255;
      int yy = ss >> 4, xx = ss & 15;
      gridh[((((nn * 8 + zz) * 16 + yy) * 16 + xx) * 12) + cd] =
          (_Float16)(c[j] + bias);
    }
  }
}

// ===== guide map + trilinear slice + per-pixel 3x4 affine (streaming) =======
#define GH   (8 * 16 * 16 * 12)   // halves per batch grid slab
#define HW   (1024 * 1024)
#define PPB  4096                 // pixels per block (256 blocks per batch)

__global__ __launch_bounds__(256)
void slice_apply_kernel(const float* __restrict__ guidance,  // (N,3,1024,1024)
                        const _Float16* __restrict__ gridh,  // (N,8,16,16,12)
                        const float* __restrict__ Mmat,      // (3,3) flat
                        const float* __restrict__ Mbias,     // (3)
                        const float* __restrict__ thrp,      // (3,16)
                        const float* __restrict__ slpp,      // (3,16)
                        const float* __restrict__ gbias,     // scalar
                        float* __restrict__ out) {           // (N,3,1024,1024)
  __shared__ __align__(16) _Float16 G[GH];   // 48 KB grid slab in LDS
  __shared__ float THR[48];
  __shared__ float SLP[48];

  const int n   = blockIdx.y;
  const int tid = threadIdx.x;

  // cooperative 16B-wide load of the grid slab into LDS
  const uint4* src4 = (const uint4*)(gridh + n * GH);
  uint4* dst4 = (uint4*)G;
  #pragma unroll
  for (int i = 0; i < GH / 8 / 256; ++i)     // 3072 uint4 / 256 threads
    dst4[tid + i * 256] = src4[tid + i * 256];
  if (tid < 48) { THR[tid] = thrp[tid]; SLP[tid] = slpp[tid]; }
  __syncthreads();

  // uniform params
  float m00 = Mmat[0], m01 = Mmat[1], m02 = Mmat[2];
  float m10 = Mmat[3], m11 = Mmat[4], m12 = Mmat[5];
  float m20 = Mmat[6], m21 = Mmat[7], m22 = Mmat[8];
  float mb0 = Mbias[0], mb1 = Mbias[1], mb2 = Mbias[2];
  float gb  = gbias[0];

  const float* pR = guidance + (size_t)(n * 3 + 0) * HW;
  const float* pG = guidance + (size_t)(n * 3 + 1) * HW;
  const float* pB = guidance + (size_t)(n * 3 + 2) * HW;
  float* oR = out + (size_t)(n * 3 + 0) * HW;
  float* oG = out + (size_t)(n * 3 + 1) * HW;
  float* oB = out + (size_t)(n * 3 + 2) * HW;

  const int base = blockIdx.x * PPB;
  for (int i = tid; i < PPB; i += 256) {
    const int pix = base + i;
    const int y = pix >> 10, x = pix & 1023;

    float gr = pR[pix], gg = pG[pix], gb_ = pB[pix];

    // color matrix (g @ M + bias)
    float c0 = fmaf(gr, m00, fmaf(gg, m10, fmaf(gb_, m20, mb0)));
    float c1 = fmaf(gr, m01, fmaf(gg, m11, fmaf(gb_, m21, mb1)));
    float c2 = fmaf(gr, m02, fmaf(gg, m12, fmaf(gb_, m22, mb2)));

    // per-channel piecewise-linear curves
    float acc = 0.0f;
    #pragma unroll
    for (int c = 0; c < 3; ++c) {
      float v = (c == 0) ? c0 : (c == 1) ? c1 : c2;
      #pragma unroll
      for (int t = 0; t < 16; ++t)
        acc = fmaf(SLP[c * 16 + t], fmaxf(v - THR[c * 16 + t], 0.0f), acc);
    }
    float guide = fminf(fmaxf(acc * (1.0f / 3.0f) + gb, 0.0f), 1.0f);

    // trilinear sample coords (align_corners=False, border clamp)
    float ix = x * (16.0f / 1023.0f) - 0.5f;
    float iy = y * (16.0f / 1023.0f) - 0.5f;
    float iz = guide * 8.0f - 0.5f;
    float x0f = floorf(ix), y0f = floorf(iy), z0f = floorf(iz);
    float wx = ix - x0f, wy = iy - y0f, wz = iz - z0f;
    int x0 = min(max((int)x0f, 0), 15), x1 = min(max((int)x0f + 1, 0), 15);
    int y0 = min(max((int)y0f, 0), 15), y1 = min(max((int)y0f + 1, 0), 15);
    int z0 = min(max((int)z0f, 0), 7),  z1 = min(max((int)z0f + 1, 0), 7);

    float co[12];
    #pragma unroll
    for (int c = 0; c < 12; ++c) co[c] = 0.0f;
    #pragma unroll
    for (int dz = 0; dz < 2; ++dz) {
      int zi = dz ? z1 : z0;
      float wzf = dz ? wz : (1.0f - wz);
      #pragma unroll
      for (int dy = 0; dy < 2; ++dy) {
        int yi = dy ? y1 : y0;
        float wyf = dy ? wy : (1.0f - wy);
        #pragma unroll
        for (int dx = 0; dx < 2; ++dx) {
          int xi = dx ? x1 : x0;
          float w = wzf * wyf * (dx ? wx : (1.0f - wx));
          const _Float16* p = G + ((zi * 16 + yi) * 16 + xi) * 12;
          #pragma unroll
          for (int c = 0; c < 12; ++c) co[c] = fmaf(w, (float)p[c], co[c]);
        }
      }
    }

    oR[pix] = fmaf(gr, co[0], fmaf(gg, co[1],  fmaf(gb_, co[2],  co[3])));
    oG[pix] = fmaf(gr, co[4], fmaf(gg, co[5],  fmaf(gb_, co[6],  co[7])));
    oB[pix] = fmaf(gr, co[8], fmaf(gg, co[9],  fmaf(gb_, co[10], co[11])));
  }
}

// ============================ host launcher =================================
extern "C" void kernel_launch(void* const* d_in, const int* in_sizes, int n_in,
                              void* d_out, int out_size, void* d_ws, size_t ws_size,
                              hipStream_t stream) {
  const float* z        = (const float*)d_in[0];
  const float* guidance = (const float*)d_in[1];
  const float* w1 = (const float*)d_in[2];
  const float* b1 = (const float*)d_in[3];
  const float* w2 = (const float*)d_in[4];
  const float* b2 = (const float*)d_in[5];
  const float* w3 = (const float*)d_in[6];
  const float* b3 = (const float*)d_in[7];
  const float* w4 = (const float*)d_in[8];
  const float* b4 = (const float*)d_in[9];
  const float* M     = (const float*)d_in[10];
  const float* Mb    = (const float*)d_in[11];
  const float* thr   = (const float*)d_in[12];
  const float* slp   = (const float*)d_in[13];
  const float* gbias = (const float*)d_in[14];

  const int N = in_sizes[0] / (4 * 64 * 64);

  // workspace layout
  float*    x1    = (float*)d_ws;                       // N*16*32*32 f32
  float*    x2    = x1 + (size_t)N * 16 * 32 * 32;      // N*32*16*16 f32
  _Float16* gridh = (_Float16*)(x2 + (size_t)N * 32 * 16 * 16); // N*24576 f16

  const int t1 = N * 16 * 32 * 32;
  conv1_kernel<<<(t1 + 255) / 256, 256, 0, stream>>>(z, w1, b1, x1, t1);

  const int t2 = N * 32 * 16 * 16;
  conv2_kernel<<<(t2 + 255) / 256, 256, 0, stream>>>(x1, w2, b2, x2, t2);

  coeff_gemm_kernel<<<N * 16, 32, 0, stream>>>(x2, w3, b3, w4, b4, gridh);

  slice_apply_kernel<<<dim3(HW / PPB, N), 256, 0, stream>>>(
      guidance, gridh, M, Mb, thr, slp, gbias, (float*)d_out);
}
